// ResNetLinLnDrop_6803228196916
// MI455X (gfx1250) — compile-verified, matrix operand order checked by reading, not run on hardware
//
#include <hip/hip_runtime.h>
#include <hip/hip_bf16.h>

// ---------------------------------------------------------------------------
// MI455X fused residual MLP block:
//   LN(768) -> Linear 768->384 -> Swish -> LN(384) -> Linear 384->768 -> Swish -> +x
// f16 WMMA (v_wmma_f32_16x16x32_f16) with f32 accumulate; intermediates in LDS.
// 2x2 register blocking per wave: 4 WMMAs per 2 A-frag + 2 B-frag loads.
// ---------------------------------------------------------------------------

typedef _Float16 v4h  __attribute__((ext_vector_type(4)));
typedef _Float16 v8h  __attribute__((ext_vector_type(8)));
typedef _Float16 v16h __attribute__((ext_vector_type(16)));
typedef float    v8f  __attribute__((ext_vector_type(8)));

#define D_IN  768
#define H_MID 384
#define TM    64
#define XN_STRIDE (D_IN + 8)    // 776 halves/row: 4-dword bank rotation per row
#define H_STRIDE  (H_MID + 8)   // 392 halves/row: 4-dword bank rotation per row

#define WMMA_F16(A, B, C) \
  __builtin_amdgcn_wmma_f32_16x16x32_f16(false, (A), false, (B), (short)0, (C), false, false)

// fast swish: x * sigmoid(x) = x * rcp(1 + exp(-x));  v_exp_f32 + v_rcp_f32 only
__device__ __forceinline__ float fast_swish(float v) {
  return v * __builtin_amdgcn_rcpf(1.f + __expf(-v));
}

// store one 16x16 f32 accumulator tile into LDS as swished f16
// C layout: lane l holds N = l&15, M = (l>>4)*8 + i
__device__ __forceinline__ void store_swish_tile(_Float16* p, int stride, const v8f& acc) {
  #pragma unroll
  for (int i = 0; i < 8; ++i) p[i * stride] = (_Float16)fast_swish(acc[i]);
}

// ---- one-time (per launch) weight conversion f32 -> f16 into workspace ----
__global__ void convert_weights(const float* __restrict__ W1,
                                const float* __restrict__ W2,
                                _Float16* __restrict__ W1h,
                                _Float16* __restrict__ W2h) {
  int i = blockIdx.x * blockDim.x + threadIdx.x;
  const int n = H_MID * D_IN;   // 294912 elements in each weight
  if (i < n) {
    W1h[i] = (_Float16)W1[i];   // layout [H][D] (row = output h, contiguous K)
    W2h[i] = (_Float16)W2[i];   // layout [D][H] (row = output d, contiguous K)
  }
}

__global__ __launch_bounds__(256)
void fused_mlp(const float* __restrict__ x,
               const _Float16* __restrict__ W1h,
               const _Float16* __restrict__ W2h,
               const float* __restrict__ g1, const float* __restrict__ b1,
               const float* __restrict__ g2, const float* __restrict__ b2,
               float* __restrict__ out)
{
  extern __shared__ __align__(16) char smem[];
  _Float16* s_xn = (_Float16*)smem;                    // TM * XN_STRIDE (f16 normalized x)
  _Float16* s_h  = s_xn + TM * XN_STRIDE;              // TM * H_STRIDE  (f16 hidden)
  float* s_sum = (float*)(s_h + TM * H_STRIDE);        // TM*4 partial sums
  float* s_sq  = s_sum + TM * 4;                       // TM*4 partial sumsq
  float* s_mu  = s_sq  + TM * 4;                       // TM
  float* s_rs  = s_mu  + TM;                           // TM

  const int  tid  = threadIdx.x;
  const long row0 = (long)blockIdx.x * TM;

  // -------- Phase A: LayerNorm over D=768, write f16 normalized tile --------
  {
    const int row = tid >> 2, q = tid & 3;             // 4 threads per row
    const float4* x4 = (const float4*)(x + (row0 + row) * (long)D_IN + q * 192);
    float sum = 0.f, sq = 0.f;
    #pragma unroll 4
    for (int j = 0; j < 48; ++j) {
      float4 v = x4[j];
      sum += v.x + v.y + v.z + v.w;
      sq  += v.x*v.x + v.y*v.y + v.z*v.z + v.w*v.w;
    }
    s_sum[row*4+q] = sum; s_sq[row*4+q] = sq;
  }
  __syncthreads();
  if (tid < TM) {
    float sum = s_sum[tid*4] + s_sum[tid*4+1] + s_sum[tid*4+2] + s_sum[tid*4+3];
    float sq  = s_sq [tid*4] + s_sq [tid*4+1] + s_sq [tid*4+2] + s_sq [tid*4+3];
    float mu  = sum * (1.f / D_IN);
    float var = sq  * (1.f / D_IN) - mu * mu;
    s_mu[tid] = mu;
    s_rs[tid] = __frsqrt_rn(var + 1e-5f);
  }
  __syncthreads();
  {
    const int row = tid >> 2, q = tid & 3;
    const float mu = s_mu[row], rs = s_rs[row];
    const float4* x4 = (const float4*)(x + (row0 + row) * (long)D_IN + q * 192);
    const float4* g4 = (const float4*)(g1 + q * 192);
    const float4* bb4 = (const float4*)(b1 + q * 192);
    _Float16* dst = s_xn + row * XN_STRIDE + q * 192;
    #pragma unroll 4
    for (int j = 0; j < 48; ++j) {
      float4 v = x4[j], g = g4[j], bb = bb4[j];
      v4h o;
      o[0] = (_Float16)((v.x - mu) * rs * g.x + bb.x);
      o[1] = (_Float16)((v.y - mu) * rs * g.y + bb.y);
      o[2] = (_Float16)((v.z - mu) * rs * g.z + bb.z);
      o[3] = (_Float16)((v.w - mu) * rs * g.w + bb.w);
      *(v4h*)(dst + j * 4) = o;
    }
  }
  __syncthreads();

  // wave/lane geometry: 8 waves = 2 row-pairs (32 rows) x 4 column quarters
  const int wave = tid >> 5, lane = tid & 31;
  const int rp = wave & 1;                             // rows rp*32 .. rp*32+31
  const int qc = wave >> 1;                            // column quarter 0..3
  const int ln = lane & 15, lk = lane >> 4;            // lane col / K-half select

  // -------- Phase B: GEMM1 [64x768] x W1^T [768x384] -> swish -> s_h --------
  {
    const _Float16* a0p = s_xn + (rp * 32 + ln) * XN_STRIDE + lk * 16;
    const _Float16* a1p = a0p + 16 * XN_STRIDE;
    for (int cp = 0; cp < 3; ++cp) {
      const int c0 = (qc * 6 + cp * 2) * 16;           // base col of tile pair
      const _Float16* b0p = W1h + (c0 + ln) * (long)D_IN + lk * 16;
      const _Float16* b1p = b0p + 16 * (long)D_IN;
      v8f acc00 = {}, acc01 = {}, acc10 = {}, acc11 = {};
      #pragma unroll 2
      for (int k0 = 0; k0 < D_IN; k0 += 32) {
        union { v16h v; v8h h[2]; } a0, a1;
        a0.h[0] = *(const v8h*)(a0p + k0); a0.h[1] = *(const v8h*)(a0p + k0 + 8);
        a1.h[0] = *(const v8h*)(a1p + k0); a1.h[1] = *(const v8h*)(a1p + k0 + 8);
        v16h b0 = *(const v16h*)(b0p + k0);            // 32B-aligned global (L2-hot)
        v16h b1 = *(const v16h*)(b1p + k0);
        acc00 = WMMA_F16(a0.v, b0, acc00);
        acc01 = WMMA_F16(a0.v, b1, acc01);
        acc10 = WMMA_F16(a1.v, b0, acc10);
        acc11 = WMMA_F16(a1.v, b1, acc11);
      }
      _Float16* hp0 = s_h + (rp * 32 + lk * 8) * H_STRIDE;         // rows rp*32+lk*8..
      _Float16* hp1 = hp0 + 16 * H_STRIDE;                         // +16 rows
      store_swish_tile(hp0 + c0 + ln,      H_STRIDE, acc00);
      store_swish_tile(hp0 + c0 + 16 + ln, H_STRIDE, acc01);
      store_swish_tile(hp1 + c0 + ln,      H_STRIDE, acc10);
      store_swish_tile(hp1 + c0 + 16 + ln, H_STRIDE, acc11);
    }
  }
  __syncthreads();

  // -------- Phase C: LayerNorm over H=384 (in-place on s_h) --------
  {
    const int row = tid >> 2, q = tid & 3;
    const _Float16* hr = s_h + row * H_STRIDE + q * 96;
    float sum = 0.f, sq = 0.f;
    #pragma unroll 4
    for (int j = 0; j < 96; ++j) { float v = (float)hr[j]; sum += v; sq += v * v; }
    s_sum[row*4+q] = sum; s_sq[row*4+q] = sq;
  }
  __syncthreads();
  if (tid < TM) {
    float sum = s_sum[tid*4] + s_sum[tid*4+1] + s_sum[tid*4+2] + s_sum[tid*4+3];
    float sq  = s_sq [tid*4] + s_sq [tid*4+1] + s_sq [tid*4+2] + s_sq [tid*4+3];
    float mu  = sum * (1.f / H_MID);
    float var = sq  * (1.f / H_MID) - mu * mu;
    s_mu[tid] = mu;
    s_rs[tid] = __frsqrt_rn(var + 1e-5f);
  }
  __syncthreads();
  {
    const int row = tid >> 2, q = tid & 3;
    const float mu = s_mu[row], rs = s_rs[row];
    _Float16* hr = s_h + row * H_STRIDE + q * 96;
    const float* g = g2 + q * 96;
    const float* bb = b2 + q * 96;
    #pragma unroll 4
    for (int j = 0; j < 96; ++j) {
      float v = (float)hr[j];
      hr[j] = (_Float16)((v - mu) * rs * g[j] + bb[j]);
    }
  }
  __syncthreads();

  // -------- Phase D: GEMM2 [64x384] x W2^T [384x768] -> swish -> +x -> out --------
  {
    const _Float16* a0p = s_h + (rp * 32 + ln) * H_STRIDE + lk * 16;
    const _Float16* a1p = a0p + 16 * H_STRIDE;
    for (int cp = 0; cp < 6; ++cp) {
      const int c0 = (qc * 12 + cp * 2) * 16;
      const _Float16* b0p = W2h + (c0 + ln) * (long)H_MID + lk * 16;
      const _Float16* b1p = b0p + 16 * (long)H_MID;
      v8f acc00 = {}, acc01 = {}, acc10 = {}, acc11 = {};
      #pragma unroll 2
      for (int k0 = 0; k0 < H_MID; k0 += 32) {
        union { v16h v; v8h h[2]; } a0, a1;
        a0.h[0] = *(const v8h*)(a0p + k0); a0.h[1] = *(const v8h*)(a0p + k0 + 8);
        a1.h[0] = *(const v8h*)(a1p + k0); a1.h[1] = *(const v8h*)(a1p + k0 + 8);
        v16h b0 = *(const v16h*)(b0p + k0);
        v16h b1 = *(const v16h*)(b1p + k0);
        acc00 = WMMA_F16(a0.v, b0, acc00);
        acc01 = WMMA_F16(a0.v, b1, acc01);
        acc10 = WMMA_F16(a1.v, b0, acc10);
        acc11 = WMMA_F16(a1.v, b1, acc11);
      }
      // epilogue: swish + residual + store, 4 tiles
      const long r0 = row0 + rp * 32 + lk * 8;         // tiles 00/01
      const long r1 = r0 + 16;                         // tiles 10/11
      #pragma unroll
      for (int i = 0; i < 8; ++i) {
        long i0 = (r0 + i) * (long)D_IN + c0 + ln;
        long i1 = (r1 + i) * (long)D_IN + c0 + ln;
        out[i0]      = fast_swish(acc00[i]) + x[i0];
        out[i0 + 16] = fast_swish(acc01[i]) + x[i0 + 16];
        out[i1]      = fast_swish(acc10[i]) + x[i1];
        out[i1 + 16] = fast_swish(acc11[i]) + x[i1 + 16];
      }
    }
  }
}

extern "C" void kernel_launch(void* const* d_in, const int* in_sizes, int n_in,
                              void* d_out, int out_size, void* d_ws, size_t ws_size,
                              hipStream_t stream) {
  const float* x  = (const float*)d_in[0];
  const float* W1 = (const float*)d_in[1];
  const float* W2 = (const float*)d_in[2];
  const float* g1 = (const float*)d_in[3];
  const float* b1 = (const float*)d_in[4];
  const float* g2 = (const float*)d_in[5];
  const float* b2 = (const float*)d_in[6];
  float* out = (float*)d_out;

  _Float16* W1h = (_Float16*)d_ws;
  _Float16* W2h = W1h + (size_t)H_MID * D_IN;

  const int nw = H_MID * D_IN;
  convert_weights<<<(nw + 255) / 256, 256, 0, stream>>>(W1, W2, W1h, W2h);

  const long rows = (long)in_sizes[0] / D_IN;          // 65536
  const int  grid = (int)(rows / TM);                  // 1024 blocks
  const size_t shmem = (size_t)(TM * XN_STRIDE + TM * H_STRIDE) * sizeof(_Float16)
                     + (size_t)(TM * 4 * 2 + TM * 2) * sizeof(float);  // ~152 KB
  hipFuncSetAttribute((const void*)fused_mlp,
                      hipFuncAttributeMaxDynamicSharedMemorySize, (int)shmem);
  fused_mlp<<<grid, 256, shmem, stream>>>(x, W1h, W2h, g1, b1, g2, b2, out);
}
